// SelfAttention_5188320494084
// MI455X (gfx1250) — compile-verified, matrix operand order checked by reading, not run on hardware
//
#include <hip/hip_runtime.h>
#include <hip/hip_bf16.h>
#include <math.h>

typedef __bf16 bf16_t;
typedef __attribute__((ext_vector_type(16))) __bf16 v16bf;
typedef __attribute__((ext_vector_type(8)))  float  v8f;
typedef unsigned int u32x4 __attribute__((ext_vector_type(4)));
typedef int          i32x8 __attribute__((ext_vector_type(8)));
typedef int          i32x4 __attribute__((ext_vector_type(4)));

#define DMODEL 1024
#define SEQ    2048
#define BATCH  8

#if defined(__has_builtin)
#if __has_builtin(__builtin_amdgcn_tensor_load_to_lds) && \
    __has_builtin(__builtin_amdgcn_s_wait_tensorcnt)
#define USE_TDM 1
#endif
#endif

// Load one 16x32 bf16 WMMA A/B fragment (ISA 7.12.2 layout):
// lane r=lane&15 owns matrix row r; lane half h=lane>>4 selects k-chunk.
// VGPR0-3: k = h*8 .. h*8+7 ; VGPR4-7: k = 16+h*8 .. 16+h*8+7
__device__ __forceinline__ v16bf frag_ld(const bf16_t* __restrict__ base,
                                         int stride, int lane) {
    const int r = lane & 15;
    const int h = lane >> 4;
    const bf16_t* p = base + (size_t)r * stride + h * 8;
    v16bf f;
    ((uint4*)&f)[0] = *(const uint4*)(p);        // 8 bf16: k = h*8..
    ((uint4*)&f)[1] = *(const uint4*)(p + 16);   // 8 bf16: k = 16+h*8..
    return f;
}

#ifdef USE_TDM
// TDM: DMA a 32-row x 1024-col bf16 tile (row stride 1024 elems in global)
// into LDS, iterate mode: one 2KB row per iteration, LDS row stride 1032
// halves (2064B = 516 dwords = 4 mod 64 banks -> conflict-free b128 reads).
// Descriptor per CDNA5 ISA.
// This toolchain's builtin takes 6 args: (g0, g1, g2, g3, g4, cpol).
__device__ __forceinline__ void tdm_load_k_tile(unsigned lds_off_bytes,
                                                const bf16_t* gsrc) {
    unsigned long long ga = (unsigned long long)(uintptr_t)gsrc;
    u32x4 g0;
    g0.x = 1u;                                   // count=1 (valid user desc)
    g0.y = lds_off_bytes;                        // lds_addr
    g0.z = (unsigned)(ga & 0xffffffffu);         // global_addr[31:0]
    g0.w = (unsigned)((ga >> 32) & 0x01ffffffu)  // global_addr[56:32]
         | (2u << 30);                           // type=2 ("image")

    const unsigned td0 = 1024, td1 = 2048;       // tensor dims (elements)
    const unsigned tile0 = 1024, tile1 = 1;      // 1 row per iteration
    const unsigned stride0 = 1024;               // dim0 stride (elements)
    i32x8 g1;
    g1[0] = (int)((1u << 16) | (1u << 19));      // data_size=2B, iterate_en=1
    g1[1] = (int)((td0 & 0xffffu) << 16);                          // dim0 lo
    g1[2] = (int)(((td0 >> 16) & 0xffffu) | ((td1 & 0xffffu) << 16));
    g1[3] = (int)(((td1 >> 16) & 0xffffu) | ((tile0 & 0xffffu) << 16));
    g1[4] = (int)(tile1 & 0xffffu);              // tile_dim1, tile_dim2=0
    g1[5] = (int)stride0;                        // dim0_stride[31:0]
    g1[6] = 0;
    g1[7] = 0;

    i32x4 g2;                                    // iterate-mode group 2
    g2[0] = 1;                                   // tensor_dim2 (benign)
    g2[1] = 1032;                                // lds_addr_increment (elems)
    g2[2] = 1024;                                // global_addr_increment
    g2[3] = (int)(31u << 16);                    // iterate_count: 32 rows
    i32x4 g3 = {0, 0, 0, 0};
    i32x8 g4 = {0, 0, 0, 0, 0, 0, 0, 0};         // unused trailing group
    __builtin_amdgcn_tensor_load_to_lds(g0, g1, g2, g3, g4, 0);
}
#endif

// ---------------------------------------------------------------------------
// Kernel 1: C = X @ W + b, bf16 out. TRANSPOSE_OUT=1 stores Vt[b][n][s].
// Tile: 128(M) x 128(N), K-step 32. 8 waves: 4 m-blocks x 2 n-blocks.
// ---------------------------------------------------------------------------
template <int TRANSPOSE_OUT>
__launch_bounds__(256, 2)
__global__ void qkv_gemm_kernel(const float* __restrict__ X,     // [B*S, D]
                                const float* __restrict__ W,     // [D, D]
                                const float* __restrict__ bias,  // [D]
                                bf16_t* __restrict__ out) {
    __shared__ __align__(16) bf16_t a_lds[128][40];   // [m][k]
    __shared__ __align__(16) bf16_t bt_lds[128][40];  // [n][k] (W transposed)

    const int tid  = threadIdx.x;
    const int lane = tid & 31;
    const int wid  = tid >> 5;
    const int m0   = blockIdx.x * 128;
    const int n0   = blockIdx.y * 128;
    const int wm   = wid >> 1;
    const int wn   = wid & 1;

    v8f acc[2][4];
    #pragma unroll
    for (int i = 0; i < 2; i++)
        #pragma unroll
        for (int j = 0; j < 4; j++) acc[i][j] = v8f{};

    for (int k0 = 0; k0 < DMODEL; k0 += 32) {
        #pragma unroll
        for (int t = 0; t < 4; t++) {
            int q   = tid + t * 256;
            int row = q >> 3;
            int col = (q & 7) * 4;
            const float* src = X + (size_t)(m0 + row) * DMODEL + k0 + col;
            float4 v = *(const float4*)src;
            if (k0 + 32 < DMODEL) __builtin_prefetch(src + 32, 0, 3);
            bf16_t* d = &a_lds[row][col];
            d[0] = (bf16_t)v.x; d[1] = (bf16_t)v.y;
            d[2] = (bf16_t)v.z; d[3] = (bf16_t)v.w;
        }
        #pragma unroll
        for (int t = 0; t < 4; t++) {
            int q = tid + t * 256;
            int k = q >> 5;
            int n = (q & 31) * 4;
            const float* src = W + (size_t)(k0 + k) * DMODEL + n0 + n;
            float4 v = *(const float4*)src;
            if (k0 + 32 < DMODEL) __builtin_prefetch(src + 32 * DMODEL, 0, 3);
            bt_lds[n + 0][k] = (bf16_t)v.x; bt_lds[n + 1][k] = (bf16_t)v.y;
            bt_lds[n + 2][k] = (bf16_t)v.z; bt_lds[n + 3][k] = (bf16_t)v.w;
        }
        __syncthreads();

        v16bf afrag[2], bfrag[4];
        #pragma unroll
        for (int i = 0; i < 2; i++)
            afrag[i] = frag_ld(&a_lds[wm * 32 + i * 16][0], 40, lane);
        #pragma unroll
        for (int j = 0; j < 4; j++)
            bfrag[j] = frag_ld(&bt_lds[wn * 64 + j * 16][0], 40, lane);

        #pragma unroll
        for (int i = 0; i < 2; i++)
            #pragma unroll
            for (int j = 0; j < 4; j++)
                acc[i][j] = __builtin_amdgcn_wmma_f32_16x16x32_bf16(
                    false, afrag[i], false, bfrag[j], (short)0, acc[i][j],
                    false, false);
        __syncthreads();
    }

    const int half  = lane >> 4;
    const int nlane = lane & 15;
    #pragma unroll
    for (int j = 0; j < 4; j++) {
        int col  = n0 + wn * 64 + j * 16 + nlane;
        float bb = bias[col];
        #pragma unroll
        for (int i = 0; i < 2; i++) {
            #pragma unroll
            for (int e = 0; e < 8; e++) {
                int   row = m0 + wm * 32 + i * 16 + half * 8 + e;
                float val = acc[i][j][e] + bb;
                if (TRANSPOSE_OUT) {
                    int b = row >> 11;
                    int s = row & 2047;
                    out[((size_t)b * DMODEL + col) * SEQ + s] = (bf16_t)val;
                } else {
                    out[(size_t)row * DMODEL + col] = (bf16_t)val;
                }
            }
        }
    }
}

// ---------------------------------------------------------------------------
// Kernel 2: causal flash attention. Block = (32 queries, batch), 8 waves,
// each wave owns a 32x128 slice of O. K-tiles DMA'd by the Tensor Data
// Mover into double-buffered LDS (overlapped with compute); Vt fragments
// direct from global (L2-resident) with prefetch of the next key tile.
// Softmax is fully parallel: 8 lanes per row, butterfly shfl reductions.
// ---------------------------------------------------------------------------
__launch_bounds__(256, 1)
__global__ void attn_kernel(const bf16_t* __restrict__ Qg,  // [B*S, D]
                            const bf16_t* __restrict__ Kg,  // [B*S, D]
                            const bf16_t* __restrict__ Vt,  // [B, D, S]
                            float* __restrict__ Y) {        // [B, S, D]
    __shared__ __align__(16) float  sp[2][32][32];
    __shared__ __align__(16) bf16_t p_lds[32][40];
    __shared__ float row_m[32], row_l[32], row_a[32];
#ifdef USE_TDM
    __shared__ __align__(16) bf16_t k_lds[2][32 * 1032];  // row stride 1032
#endif

    const int tid  = threadIdx.x;
    const int lane = tid & 31;
    const int wid  = tid >> 5;
    const int qt   = blockIdx.x;
    const int b    = blockIdx.y;
    const int m0   = qt * 32;

    const bf16_t* qb = Qg + (size_t)b * SEQ * DMODEL;
    const bf16_t* kb = Kg + (size_t)b * SEQ * DMODEL;
    const bf16_t* vb = Vt + (size_t)b * DMODEL * SEQ;

    const int mh  = wid & 1;
    const int th  = (wid >> 1) & 1;
    const int dh  = wid >> 2;
    const int dv0 = wid * 128;

    v8f acc[2][8];
    #pragma unroll
    for (int i = 0; i < 2; i++)
        #pragma unroll
        for (int j = 0; j < 8; j++) acc[i][j] = v8f{};

    if (tid < 32) { row_m[tid] = -3.0e38f; row_l[tid] = 0.0f; }
    __syncthreads();

    const float scale  = 0.03125f;  // 1/sqrt(1024)
    const int   nsteps = qt + 1;    // causal: skip fully-masked key tiles

#ifdef USE_TDM
    if (wid == 0)
        tdm_load_k_tile((unsigned)(uintptr_t)&k_lds[0][0], kb);
#endif

    for (int st = 0; st < nsteps; ++st) {
        const int t0 = st * 32;

#ifdef USE_TDM
        const int cb = st & 1;
        if (wid == 0) __builtin_amdgcn_s_wait_tensorcnt(0);
        __syncthreads();  // publish DMA'd K tile to all waves
        if (wid == 0 && st + 1 < nsteps)
            tdm_load_k_tile((unsigned)(uintptr_t)&k_lds[(st + 1) & 1][0],
                            kb + (size_t)(t0 + 32) * DMODEL);
#endif

        // ---- phase 1: partial scores S = Q K^T over this wave's d-half ----
        v8f s = v8f{};
        #pragma unroll 4
        for (int kk = 0; kk < 16; ++kk) {
            int d0 = dh * 512 + kk * 32;
            v16bf aq = frag_ld(qb + (size_t)(m0 + mh * 16) * DMODEL + d0,
                               DMODEL, lane);
#ifdef USE_TDM
            v16bf bk = frag_ld(&k_lds[cb][(size_t)(th * 16) * 1032 + d0],
                               1032, lane);
#else
            v16bf bk = frag_ld(kb + (size_t)(t0 + th * 16) * DMODEL + d0,
                               DMODEL, lane);
#endif
            s = __builtin_amdgcn_wmma_f32_16x16x32_bf16(
                false, aq, false, bk, (short)0, s, false, false);
        }
        {
            const int half = lane >> 4, nl = lane & 15;
            #pragma unroll
            for (int e = 0; e < 8; e++)
                sp[dh][mh * 16 + half * 8 + e][th * 16 + nl] = s[e];
        }
        __syncthreads();

        // ---- phase 2: online softmax, all 256 threads.
        //      8 lanes per row, 4 cols each; butterfly shfl reductions. ----
        {
            const int r    = tid >> 3;         // row 0..31 (4 rows per wave)
            const int c0   = (tid & 7) * 4;    // this lane's 4 columns
            const int qrow = m0 + r;
            float sc[4];
            float lmax = -3.0e38f;
            #pragma unroll
            for (int u = 0; u < 4; u++) {
                int c = c0 + u;
                float v = (sp[0][r][c] + sp[1][r][c]) * scale;
                if (t0 + c > qrow) v = -3.0e38f;   // causal mask
                sc[u] = v;
                lmax  = fmaxf(lmax, v);
            }
            #pragma unroll
            for (int off = 1; off < 8; off <<= 1)
                lmax = fmaxf(lmax, __shfl_xor(lmax, off, 32));

            float mx = row_m[r];
            float nm = fmaxf(mx, lmax);
            float lsum = 0.0f;
            bf16_t pv[4];
            #pragma unroll
            for (int u = 0; u < 4; u++) {
                float p = __expf(sc[u] - nm);
                pv[u] = (bf16_t)p;
                lsum += p;
            }
            #pragma unroll
            for (int off = 1; off < 8; off <<= 1)
                lsum += __shfl_xor(lsum, off, 32);

            *(unsigned long long*)&p_lds[r][c0] =
                *(unsigned long long*)pv;          // 4 bf16 as one b64 store
            if ((tid & 7) == 0) {
                float alpha = __expf(mx - nm);
                row_a[r] = alpha;
                row_l[r] = row_l[r] * alpha + lsum;
                row_m[r] = nm;
            }
        }
        __syncthreads();

        // ---- phase 3: O = O*alpha + P @ Vt (wave's 32x128 slice) ----
        {
            const int half = lane >> 4;
            float al[2][8];
            #pragma unroll
            for (int i = 0; i < 2; i++)
                #pragma unroll
                for (int e = 0; e < 8; e++)
                    al[i][e] = row_a[i * 16 + half * 8 + e];
            #pragma unroll
            for (int i = 0; i < 2; i++)
                #pragma unroll
                for (int j = 0; j < 8; j++)
                    #pragma unroll
                    for (int e = 0; e < 8; e++)
                        acc[i][j][e] *= al[i][e];

            v16bf ap0 = frag_ld(&p_lds[0][0],  40, lane);
            v16bf ap1 = frag_ld(&p_lds[16][0], 40, lane);
            #pragma unroll
            for (int j = 0; j < 8; j++) {
                int dv = dv0 + j * 16;
                const bf16_t* vsrc = vb + (size_t)dv * SEQ + t0;
                v16bf bv = frag_ld(vsrc, SEQ, lane);
                if (st + 1 < nsteps) __builtin_prefetch(vsrc + 32, 0, 3);
                acc[0][j] = __builtin_amdgcn_wmma_f32_16x16x32_bf16(
                    false, ap0, false, bv, (short)0, acc[0][j], false, false);
                acc[1][j] = __builtin_amdgcn_wmma_f32_16x16x32_bf16(
                    false, ap1, false, bv, (short)0, acc[1][j], false, false);
            }
        }
        __syncthreads();
    }

    // ---- epilogue: normalize by row sums and store f32 ----
    const int half = lane >> 4, nl = lane & 15;
    float inv[2][8];
    #pragma unroll
    for (int i = 0; i < 2; i++)
        #pragma unroll
        for (int e = 0; e < 8; e++)
            inv[i][e] = 1.0f / row_l[i * 16 + half * 8 + e];
    #pragma unroll
    for (int i = 0; i < 2; i++)
        #pragma unroll
        for (int j = 0; j < 8; j++)
            #pragma unroll
            for (int e = 0; e < 8; e++) {
                int row = m0 + i * 16 + half * 8 + e;
                int col = dv0 + j * 16 + nl;
                Y[((size_t)b * SEQ + row) * DMODEL + col] =
                    acc[i][j][e] * inv[i][e];
            }
}

// ---------------------------------------------------------------------------
extern "C" void kernel_launch(void* const* d_in, const int* in_sizes, int n_in,
                              void* d_out, int out_size, void* d_ws,
                              size_t ws_size, hipStream_t stream) {
    (void)in_sizes; (void)n_in; (void)out_size; (void)ws_size;
    const float* query = (const float*)d_in[0];
    const float* key_t = (const float*)d_in[1];
    const float* value = (const float*)d_in[2];
    const float* Wq = (const float*)d_in[3];
    const float* bq = (const float*)d_in[4];
    const float* Wk = (const float*)d_in[5];
    const float* bk = (const float*)d_in[6];
    const float* Wv = (const float*)d_in[7];
    const float* bv = (const float*)d_in[8];
    // d_in[9] = causal mask: recomputed analytically in-kernel
    float* out = (float*)d_out;

    const size_t elems = (size_t)BATCH * SEQ * DMODEL;
    bf16_t* qw  = (bf16_t*)d_ws;          // Q  bf16 [B*S, D]
    bf16_t* kw  = qw + elems;             // K  bf16 [B*S, D]
    bf16_t* vtw = kw + elems;             // Vt bf16 [B, D, S]

    dim3 gA(BATCH * SEQ / 128, DMODEL / 128);
    qkv_gemm_kernel<0><<<gA, 256, 0, stream>>>(query, Wq, bq, qw);
    qkv_gemm_kernel<0><<<gA, 256, 0, stream>>>(key_t, Wk, bk, kw);
    qkv_gemm_kernel<1><<<gA, 256, 0, stream>>>(value, Wv, bv, vtw);

    dim3 gB(SEQ / 32, BATCH);
    attn_kernel<<<gB, 256, 0, stream>>>(qw, kw, vtw, out);
}
